// Attention_4432406249902
// MI455X (gfx1250) — matrix operand order repaired and checked by reading, not compiled, not run on hardware
//
#include <hip/hip_runtime.h>

// ---------------------------------------------------------------------------
// Problem constants (from reference): B=2, P=2048, M=1024, N=16 heads, H=64
// ---------------------------------------------------------------------------
constexpr int kB = 2;
constexpr int kP = 2048;
constexpr int kM = 1024;
constexpr int kN = 16;
constexpr int kH = 64;
constexpr int kNH = kN * kH;          // 1024
constexpr float kScale = 0.125f;      // 1/sqrt(64)
constexpr float kNeg = 1.0e10f;

typedef __attribute__((ext_vector_type(16))) __bf16 v16bf;
typedef __attribute__((ext_vector_type(8)))  __bf16 v8bf;
typedef __attribute__((ext_vector_type(8)))  float  v8f;
typedef __attribute__((ext_vector_type(4)))  unsigned int v4u;
typedef __attribute__((ext_vector_type(8)))  int v8i;
typedef __attribute__((ext_vector_type(4)))  int v4i;

union V16 { v16bf v; v8bf h[2]; };

static __device__ __forceinline__ v8f wmma_bf16(v16bf a, v16bf b, v8f c) {
  // D(16x16,f32) = A(16x32,bf16) * B(32x16,bf16) + C
  return __builtin_amdgcn_wmma_f32_16x16x32_bf16(
      /*neg_a=*/false, a, /*neg_b=*/false, b,
      /*c_mod=*/(short)0, c, /*reuse_a=*/false, /*reuse_b=*/false);
}

// ---------------------------------------------------------------------------
// Tensor Data Mover: build a 2D-tile D# (groups 0/1) for bf16 data.
// Group0: [1:0]=count=1, [63:32]=lds_addr, [120:64]=global_addr, [127:126]=2.
// Group1: [17:16]=data_size(1->2B), [79:48]=tensor_dim0, [111:80]=tensor_dim1,
//         [127:112]=tile_dim0, [143:128]=tile_dim1, [207:160]=dim0_stride.
// ---------------------------------------------------------------------------
struct TdmDesc { v4u g0; v8i g1; };

static __device__ __forceinline__ TdmDesc
make_tdm_2d_bf16(unsigned lds_off, const void* gaddr,
                 unsigned tensor_d0, unsigned tensor_d1,
                 unsigned tile_d0, unsigned tile_d1,
                 unsigned long long stride0 /* elements */) {
  TdmDesc d;
  const unsigned long long ga = (unsigned long long)gaddr;
  d.g0[0] = 1u;                                              // count=1
  d.g0[1] = lds_off;                                         // lds_addr
  d.g0[2] = (unsigned)(ga & 0xffffffffu);                    // global_addr lo
  d.g0[3] = (unsigned)((ga >> 32) & 0x01ffffffu) | (2u << 30);  // hi + type=2
  const unsigned w0 = 1u << 16;                              // data_size = 2B
  const unsigned w1 = (tensor_d0 & 0xffffu) << 16;           // dim0[15:0]
  const unsigned w2 = ((tensor_d0 >> 16) & 0xffffu) | ((tensor_d1 & 0xffffu) << 16);
  const unsigned w3 = ((tensor_d1 >> 16) & 0xffffu) | ((tile_d0 & 0xffffu) << 16);
  const unsigned w4 = tile_d1 & 0xffffu;                     // tile_dim2 = 0
  const unsigned w5 = (unsigned)(stride0 & 0xffffffffu);     // dim0_stride lo
  const unsigned w6 = (unsigned)((stride0 >> 32) & 0xffffu); // dim0_stride hi
  d.g1 = v8i{(int)w0, (int)w1, (int)w2, (int)w3, (int)w4, (int)w5, (int)w6, 0};
  return d;
}

static __device__ __forceinline__ void tdm_load(const TdmDesc& d) {
  const v4i z4 = {0, 0, 0, 0};
#if defined(__clang_major__) && (__clang_major__ >= 23)
  const v8i z8 = {0, 0, 0, 0, 0, 0, 0, 0};
  __builtin_amdgcn_tensor_load_to_lds(d.g0, d.g1, z4, z4, z8, 0);
#else
  __builtin_amdgcn_tensor_load_to_lds(d.g0, d.g1, z4, z4, 0);
#endif
}

// ---------------------------------------------------------------------------
// Pack: f32 -> bf16 straight copy
// ---------------------------------------------------------------------------
__global__ void cvt_f32_bf16(const float* __restrict__ in,
                             __bf16* __restrict__ out, int n) {
  int i = blockIdx.x * blockDim.x + threadIdx.x;
  if (i < n) out[i] = (__bf16)in[i];
}

// Pack: in[g][r][c] f32 -> out[g][c][r] bf16   (per-head weight transpose)
__global__ void transpose_cvt(const float* __restrict__ in,
                              __bf16* __restrict__ out, int R, int C) {
  int g = blockIdx.y;
  int i = blockIdx.x * blockDim.x + threadIdx.x;
  if (i < R * C) {
    int r = i / C, c = i % C;
    out[(size_t)g * R * C + (size_t)c * R + r] =
        (__bf16)in[(size_t)g * R * C + (size_t)r * C + c];
  }
}

// ---------------------------------------------------------------------------
// QKV projection: q[b,p,n,h] = sum_m x[b,p,m] * W[n,m,h] + bias[n,h]
// One wave computes a 16(p) x 64(h) tile for one (b,n).  The 16xM x-tile is
// DMA'd into LDS once by the TDM and reused by all three projections; the
// weight B-tiles are register double-buffered so WMMAs overlap the loads.
// ---------------------------------------------------------------------------
__global__ void qkv_project(const __bf16* __restrict__ xbf,
                            const __bf16* __restrict__ Wqt,
                            const __bf16* __restrict__ Wkt,
                            const __bf16* __restrict__ Wvt,
                            const float* __restrict__ bq,
                            const float* __restrict__ bk,
                            const float* __restrict__ bv,
                            __bf16* __restrict__ Qb,
                            __bf16* __restrict__ Kb,
                            __bf16* __restrict__ Vt) {
  __shared__ __align__(16) __bf16 ldsX[16 * kM];  // 32 KB x tile [p][m]

  const int lane = threadIdx.x;
  const int col  = lane & 15;   // C/D column, A row, B column
  const int hi   = lane >> 4;   // K-half selector
  const int wg = blockIdx.x;
  const int pt = wg % (kP / 16);
  const int n  = (wg / (kP / 16)) % kN;
  const int b  = wg / (kN * (kP / 16));
  const int p0 = pt * 16;

  // Stage x tile via Tensor Data Mover (16 rows x kM, row stride kM).
  TdmDesc dx = make_tdm_2d_bf16((unsigned)(unsigned long long)&ldsX[0],
                                xbf + (size_t)(b * kP + p0) * kM,
                                kM, kP, kM, 16, (unsigned long long)kM);
  tdm_load(dx);
  __builtin_amdgcn_s_wait_tensorcnt(0);

  const __bf16* xrow = &ldsX[(size_t)col * kM];  // A-operand row for this lane

  const __bf16* Wt3[3]   = {Wqt, Wkt, Wvt};
  const float*  bias3[3] = {bq, bk, bv};

  for (int w = 0; w < 3; ++w) {
    const __bf16* Wt = Wt3[w] + (size_t)n * kH * kM;  // [H][M]
    v8f acc[4] = {v8f(0.f), v8f(0.f), v8f(0.f), v8f(0.f)};

    // Prime B double-buffer (4 h-tiles of the weight matrix).
    v16bf b_cur[4];
#pragma unroll
    for (int ht = 0; ht < 4; ++ht)
      b_cur[ht] = *(const v16bf*)(Wt + (size_t)(ht * 16 + col) * kM + hi * 16);

    for (int m = 0; m < kM; m += 32) {
      // Fetch next K-step's B tiles before consuming the current ones.
      const int mn = (m + 32 < kM) ? m + 32 : m;
      v16bf b_nxt[4];
#pragma unroll
      for (int ht = 0; ht < 4; ++ht)
        b_nxt[ht] = *(const v16bf*)(Wt + (size_t)(ht * 16 + col) * kM + mn + hi * 16);

      V16 a;  // A fragment from LDS: two 16B chunks
      a.h[0] = *(const v8bf*)(xrow + m + hi * 16);
      a.h[1] = *(const v8bf*)(xrow + m + hi * 16 + 8);
#pragma unroll
      for (int ht = 0; ht < 4; ++ht)
        acc[ht] = wmma_bf16(a.v, b_cur[ht], acc[ht]);
#pragma unroll
      for (int ht = 0; ht < 4; ++ht) b_cur[ht] = b_nxt[ht];
    }

    const size_t head = (size_t)(b * kN + n);
#pragma unroll
    for (int ht = 0; ht < 4; ++ht) {
      const int h = ht * 16 + col;
      const float bb = bias3[w][n * kH + h];
      if (w == 2) {
        // V transposed [H,P]: lane's 8 row-values are contiguous in p
        v8bf pk;
#pragma unroll
        for (int v = 0; v < 8; ++v) pk[v] = (__bf16)(acc[ht][v] + bb);
        *(v8bf*)(Vt + (head * kH + h) * kP + p0 + hi * 8) = pk;
      } else {
        __bf16* dst = (w == 0 ? Qb : Kb) + (head * kP) * kH;
#pragma unroll
        for (int v = 0; v < 8; ++v)
          dst[(size_t)(p0 + v + hi * 8) * kH + h] = (__bf16)(acc[ht][v] + bb);
      }
    }
  }
}

// ---------------------------------------------------------------------------
// Flash attention per (b, n, 16-row q tile).  K/V slabs are DMA'd into LDS by
// the Tensor Data Mover (double buffered, TENSORcnt-synchronized); inverted-
// causal mask, online softmax, Z accumulated via WMMA from LDS operands.
// ---------------------------------------------------------------------------
__global__ void flash_attn(const __bf16* __restrict__ Qb,
                           const __bf16* __restrict__ Kb,
                           const __bf16* __restrict__ Vt,
                           __bf16* __restrict__ Zb) {
  __shared__ __align__(16) __bf16 ldsK[2][32 * kH];  // [pk][h]
  __shared__ __align__(16) __bf16 ldsV[2][kH * 32];  // [h][pk]
  __shared__ __align__(16) __bf16 lds_p[16 * 32];    // prob tile [q][pk]

  const int lane = threadIdx.x;
  const int col  = lane & 15;
  const int hi   = lane >> 4;
  const int wg = blockIdx.x;
  const int qt = wg % (kP / 16);
  const int n  = (wg / (kP / 16)) % kN;
  const int b  = wg / (kN * (kP / 16));
  const int q0 = qt * 16;

  const size_t head = (size_t)(b * kN + n);
  const __bf16* Kbase = Kb + head * kP * kH;  // [P][H]
  const __bf16* Vbase = Vt + head * kH * kP;  // [H][P]

  // Issue one TDM load pair (K slab 32xH, V slab Hx32) for pk-step s.
  auto issue = [&](int s) {
    const int buf = (s >> 5) & 1;
    TdmDesc dk = make_tdm_2d_bf16((unsigned)(unsigned long long)&ldsK[buf][0],
                                  Kbase + (size_t)s * kH,
                                  kH, kP, kH, 32, (unsigned long long)kH);
    tdm_load(dk);
    TdmDesc dv = make_tdm_2d_bf16((unsigned)(unsigned long long)&ldsV[buf][0],
                                  Vbase + s,
                                  kP, kH, 32, kH, (unsigned long long)kP);
    tdm_load(dv);
  };

  // Q tile (A operand, reused every iteration): two K-steps over H=64
  const __bf16* qrow = Qb + (head * kP + q0 + col) * kH;
  V16 aq[2];
#pragma unroll
  for (int kk = 0; kk < 2; ++kk) {
    aq[kk].h[0] = *(const v8bf*)(qrow + kk * 32 + hi * 16);
    aq[kk].h[1] = *(const v8bf*)(qrow + kk * 32 + hi * 16 + 8);
  }

  float mrow[8], lrow[8];
#pragma unroll
  for (int v = 0; v < 8; ++v) { mrow[v] = -3.0e38f; lrow[v] = 0.f; }
  v8f zacc[4] = {v8f(0.f), v8f(0.f), v8f(0.f), v8f(0.f)};

  issue(0);    // prefill both buffers
  issue(32);

  for (int pk0 = 0; pk0 < kP; pk0 += 32) {
    // Wait for this step's slabs (the 2 newest ops belong to the next step).
    if (pk0 + 32 < kP) __builtin_amdgcn_s_wait_tensorcnt(2);
    else               __builtin_amdgcn_s_wait_tensorcnt(0);
    const __bf16* Kt = &ldsK[(pk0 >> 5) & 1][0];
    const __bf16* Vv = &ldsV[(pk0 >> 5) & 1][0];

    // Scores: two 16x16 tiles over this 32-wide pk slab (B operand from LDS)
    v8f s[2] = {v8f(0.f), v8f(0.f)};
#pragma unroll
    for (int t = 0; t < 2; ++t) {
      v16bf bk0 = *(const v16bf*)(Kt + (size_t)(t * 16 + col) * kH + hi * 16);
      v16bf bk1 = *(const v16bf*)(Kt + (size_t)(t * 16 + col) * kH + 32 + hi * 16);
      s[t] = wmma_bf16(aq[0].v, bk0, s[t]);
      s[t] = wmma_bf16(aq[1].v, bk1, s[t]);
    }

    // Mask (keep pk > q), scale, online softmax stats
    float cmax[8];
#pragma unroll
    for (int v = 0; v < 8; ++v) {
      const int qg = q0 + v + hi * 8;
#pragma unroll
      for (int t = 0; t < 2; ++t) {
        const int pg = pk0 + t * 16 + col;
        s[t][v] = (pg > qg) ? s[t][v] * kScale : -kNeg;
      }
      cmax[v] = fmaxf(s[0][v], s[1][v]);
    }
#pragma unroll
    for (int v = 0; v < 8; ++v)
      for (int off = 1; off < 16; off <<= 1)
        cmax[v] = fmaxf(cmax[v], __shfl_xor(cmax[v], off, 32));

    float corr[8], psum[8];
#pragma unroll
    for (int v = 0; v < 8; ++v) {
      const float mnew = fmaxf(mrow[v], cmax[v]);
      corr[v] = __expf(mrow[v] - mnew);
      mrow[v] = mnew;
      s[0][v] = __expf(s[0][v] - mnew);
      s[1][v] = __expf(s[1][v] - mnew);
      psum[v] = s[0][v] + s[1][v];
    }
#pragma unroll
    for (int v = 0; v < 8; ++v) {
      for (int off = 1; off < 16; off <<= 1)
        psum[v] += __shfl_xor(psum[v], off, 32);
      lrow[v] = lrow[v] * corr[v] + psum[v];
    }

    // D-layout -> A-layout via LDS ([q][pk], 32 cols)
    __syncthreads();
#pragma unroll
    for (int t = 0; t < 2; ++t)
#pragma unroll
      for (int v = 0; v < 8; ++v)
        lds_p[(v + hi * 8) * 32 + t * 16 + col] = (__bf16)s[t][v];
    __syncthreads();

    V16 ap;
    ap.h[0] = *(const v8bf*)&lds_p[col * 32 + hi * 16];
    ap.h[1] = *(const v8bf*)&lds_p[col * 32 + hi * 16 + 8];

    // Z = Z*corr + P@V ; V slab in LDS is [h][pk], contiguous B fragments
    v16bf bv4[4];
#pragma unroll
    for (int ht = 0; ht < 4; ++ht)
      bv4[ht] = *(const v16bf*)(Vv + (size_t)(ht * 16 + col) * 32 + hi * 16);
#pragma unroll
    for (int ht = 0; ht < 4; ++ht) {
#pragma unroll
      for (int v = 0; v < 8; ++v) zacc[ht][v] *= corr[v];
      zacc[ht] = wmma_bf16(ap.v, bv4[ht], zacc[ht]);
    }

    // Refill the buffer we just consumed with the step two slabs ahead.
    if (pk0 + 64 < kP) issue(pk0 + 64);
  }

  // Finalize and store Z as [B,P,N,H] bf16
  float invl[8];
#pragma unroll
  for (int v = 0; v < 8; ++v) invl[v] = 1.0f / lrow[v];
#pragma unroll
  for (int ht = 0; ht < 4; ++ht) {
    const int h = ht * 16 + col;
#pragma unroll
    for (int v = 0; v < 8; ++v) {
      const int q = q0 + v + hi * 8;
      Zb[((size_t)(b * kP + q) * kN + n) * kH + h] = (__bf16)(zacc[ht][v] * invl[v]);
    }
  }
}

// ---------------------------------------------------------------------------
// Output projection: out[p,m] = sum_k Z[p,k] * WoT[m,k] + bias_out[m]
// Z viewed as [B*P, N*H]; WoT = kernel_out transposed to [M, N*H].
// Z tile staged via TDM into LDS; WoT B-tiles register double-buffered.
// ---------------------------------------------------------------------------
__global__ void out_project(const __bf16* __restrict__ Zb,
                            const __bf16* __restrict__ WoT,
                            const float* __restrict__ bo,
                            float* __restrict__ out) {
  __shared__ __align__(16) __bf16 ldsZ[16 * kNH];  // 32 KB z tile [p][k]

  const int lane = threadIdx.x;
  const int col  = lane & 15;
  const int hi   = lane >> 4;
  const int wg = blockIdx.x;
  const int mt = wg % (kM / 64);
  const int pt = wg / (kM / 64);
  const int p0 = pt * 16, m0 = mt * 64;

  TdmDesc dz = make_tdm_2d_bf16((unsigned)(unsigned long long)&ldsZ[0],
                                Zb + (size_t)p0 * kNH,
                                kNH, kB * kP, kNH, 16, (unsigned long long)kNH);
  tdm_load(dz);
  __builtin_amdgcn_s_wait_tensorcnt(0);

  const __bf16* zrow = &ldsZ[(size_t)col * kNH];
  v8f acc[4] = {v8f(0.f), v8f(0.f), v8f(0.f), v8f(0.f)};

  v16bf b_cur[4];
#pragma unroll
  for (int mtt = 0; mtt < 4; ++mtt)
    b_cur[mtt] = *(const v16bf*)(WoT + (size_t)(m0 + mtt * 16 + col) * kNH + hi * 16);

  for (int k = 0; k < kNH; k += 32) {
    const int kn = (k + 32 < kNH) ? k + 32 : k;
    v16bf b_nxt[4];
#pragma unroll
    for (int mtt = 0; mtt < 4; ++mtt)
      b_nxt[mtt] = *(const v16bf*)(WoT + (size_t)(m0 + mtt * 16 + col) * kNH + kn + hi * 16);

    V16 a;
    a.h[0] = *(const v8bf*)(zrow + k + hi * 16);
    a.h[1] = *(const v8bf*)(zrow + k + hi * 16 + 8);
#pragma unroll
    for (int mtt = 0; mtt < 4; ++mtt)
      acc[mtt] = wmma_bf16(a.v, b_cur[mtt], acc[mtt]);
#pragma unroll
    for (int mtt = 0; mtt < 4; ++mtt) b_cur[mtt] = b_nxt[mtt];
  }
#pragma unroll
  for (int mtt = 0; mtt < 4; ++mtt) {
    const int m = m0 + mtt * 16 + col;
    const float bb = bo[m];
#pragma unroll
    for (int v = 0; v < 8; ++v)
      out[(size_t)(p0 + v + hi * 8) * kM + m] = acc[mtt][v] + bb;
  }
}

// ---------------------------------------------------------------------------
// Host launcher
// ---------------------------------------------------------------------------
extern "C" void kernel_launch(void* const* d_in, const int* in_sizes, int n_in,
                              void* d_out, int out_size, void* d_ws, size_t ws_size,
                              hipStream_t stream) {
  const float* x  = (const float*)d_in[0];
  const float* wq = (const float*)d_in[1];
  const float* bq = (const float*)d_in[2];
  const float* wk = (const float*)d_in[3];
  const float* bk = (const float*)d_in[4];
  const float* wv = (const float*)d_in[5];
  const float* bv = (const float*)d_in[6];
  const float* wo = (const float*)d_in[7];
  const float* bo = (const float*)d_in[8];
  float* out = (float*)d_out;

  char* ws = (char*)d_ws;
  size_t off = 0;
  auto salloc = [&](size_t bytes) {
    char* p = ws + off;
    off += (bytes + 255) & ~(size_t)255;
    return p;
  };
  const size_t BPM = (size_t)kB * kP * kM;          // 4M elems
  const size_t WSZ = (size_t)kN * kM * kH;          // 1M elems
  const size_t QSZ = (size_t)kB * kN * kP * kH;     // 4M elems

  __bf16* xbf = (__bf16*)salloc(BPM * 2);
  __bf16* Wqt = (__bf16*)salloc(WSZ * 2);
  __bf16* Wkt = (__bf16*)salloc(WSZ * 2);
  __bf16* Wvt = (__bf16*)salloc(WSZ * 2);
  __bf16* WoT = (__bf16*)salloc((size_t)kNH * kM * 2);
  __bf16* Qb  = (__bf16*)salloc(QSZ * 2);
  __bf16* Kb  = (__bf16*)salloc(QSZ * 2);
  __bf16* Vt  = (__bf16*)salloc(QSZ * 2);
  __bf16* Zb  = (__bf16*)salloc(BPM * 2);           // [B,P,N*H]
  (void)ws_size; (void)in_sizes; (void)n_in; (void)out_size;

  // 1) pack x
  cvt_f32_bf16<<<dim3((unsigned)((BPM + 255) / 256)), dim3(256), 0, stream>>>(
      x, xbf, (int)BPM);
  // 2) per-head weight transposes [N,M,H] -> [N,H,M]
  {
    dim3 g((kM * kH + 255) / 256, kN);
    transpose_cvt<<<g, dim3(256), 0, stream>>>(wq, Wqt, kM, kH);
    transpose_cvt<<<g, dim3(256), 0, stream>>>(wk, Wkt, kM, kH);
    transpose_cvt<<<g, dim3(256), 0, stream>>>(wv, Wvt, kM, kH);
  }
  // 3) kernel_out [N*H, M] -> WoT [M, N*H]
  transpose_cvt<<<dim3((kNH * kM + 255) / 256, 1), dim3(256), 0, stream>>>(
      wo, WoT, kNH, kM);
  // 4) QKV projection: one wave per (b,n,p-tile), x tile via TDM
  qkv_project<<<dim3(kB * kN * (kP / 16)), dim3(32), 0, stream>>>(
      xbf, Wqt, Wkt, Wvt, bq, bk, bv, Qb, Kb, Vt);
  // 5) flash attention (TDM double-buffered K/V in LDS)
  flash_attn<<<dim3(kB * kN * (kP / 16)), dim3(32), 0, stream>>>(Qb, Kb, Vt, Zb);
  // 6) output projection (Z tile via TDM)
  out_project<<<dim3((kB * kP / 16) * (kM / 64)), dim3(32), 0, stream>>>(
      Zb, WoT, bo, out);
}
